// NFDiagonalMaskedSelfAttention_54099408060577
// MI455X (gfx1250) — matrix operand order repaired and checked by reading, not compiled, not run on hardware
//
#include <hip/hip_runtime.h>
#include <hip/hip_bf16.h>

// ---------------------------------------------------------------------------
// NFDiagonalMaskedSelfAttention for MI455X (gfx1250, wave32, WMMA)
//
//  x:[4,2048,1024] f32, wq/wk/wv/wo:[1024,1024] f32 (stored [out,in])
//  out = softmax(mask(QK^T/8)) V  -> @ wo^T,  f32 out
//
// bf16 WMMA (v_wmma_f32_16x16x32_bf16) everywhere, f32 accumulate.
// K tiles staged to LDS with GLOBAL_LOAD_ASYNC_TO_LDS_B128 (double-buffered).
// ---------------------------------------------------------------------------

typedef __attribute__((ext_vector_type(16))) __bf16        v16bf;
typedef __attribute__((ext_vector_type(8)))  float         v8f;
typedef __attribute__((ext_vector_type(4)))  float         v4f;
typedef __attribute__((ext_vector_type(4)))  unsigned int  v4u;
typedef __attribute__((ext_vector_type(4)))  int           v4i;
typedef __attribute__((ext_vector_type(2)))  unsigned int  v2u;

union Frag16 {            // 16 bf16 elements = 8 VGPRs
  v16bf v;
  v4u   q[2];
  unsigned int   d[8];
  unsigned short h[16];
};
union Acc8 {              // 16x16 f32 accumulator tile (8 VGPRs / lane)
  v8f   v;
  float f[8];
};

__device__ __forceinline__ unsigned short f32_to_bf16_bits(float x) {
  unsigned u = __float_as_uint(x);
  unsigned r = u + 0x7FFFu + ((u >> 16) & 1u);   // round-to-nearest-even
  return (unsigned short)(r >> 16);
}

// ------------------------ async global->LDS staging ------------------------
#if __has_builtin(__builtin_amdgcn_global_load_async_to_lds_b128)
#define HAVE_ASYNC_LDS 1
#else
#define HAVE_ASYNC_LDS 0
#endif

__device__ __forceinline__ void async_copy16(const unsigned short* gsrc,
                                             unsigned short* lds) {
#if HAVE_ASYNC_LDS
  __builtin_amdgcn_global_load_async_to_lds_b128(
      (__attribute__((address_space(1))) v4i*)gsrc,
      (__attribute__((address_space(3))) v4i*)lds, 0, 0);
#else
  *(v4u*)lds = *(const v4u*)gsrc;
#endif
}

__device__ __forceinline__ void wait_async_lds() {
#if HAVE_ASYNC_LDS
#if __has_builtin(__builtin_amdgcn_s_wait_asynccnt)
  __builtin_amdgcn_s_wait_asynccnt(0);
#else
  asm volatile("s_wait_asynccnt 0x0" ::: "memory");
#endif
#endif
}

// -------------------------- f32 -> bf16 convert ----------------------------
__global__ void cvt_f32_bf16(const float* __restrict__ src,
                             unsigned short* __restrict__ dst, int n4) {
  int i = blockIdx.x * blockDim.x + threadIdx.x;
  if (i < n4) {
    v4f f = *(const v4f*)(src + 4l * i);
    v2u o;
    o.x = (unsigned)f32_to_bf16_bits(f.x) | ((unsigned)f32_to_bf16_bits(f.y) << 16);
    o.y = (unsigned)f32_to_bf16_bits(f.z) | ((unsigned)f32_to_bf16_bits(f.w) << 16);
    *(v2u*)(dst + 4l * i) = o;
  }
}

// ----------------------- Y = X * W^T  (bf16 WMMA) --------------------------
// X:[M,K] bf16, W:[N,K] bf16.  4 waves / block; each wave owns a 32x64
// register tile (2x4 WMMA sub-tiles) => 8 WMMA per 6 fragment loads.
// Block covers 64 rows x 128 cols.  Output: f32, bf16, or bf16-transposed.
__global__ __launch_bounds__(128)
void wmma_gemm_xwT(const unsigned short* __restrict__ X,
                   const unsigned short* __restrict__ W,
                   float* __restrict__ outF,
                   unsigned short* __restrict__ outB,
                   unsigned short* __restrict__ outBT,   // [N][M] transposed
                   int M, int N, int K)
{
  const int lane = threadIdx.x & 31;
  const int wave = threadIdx.x >> 5;   // 0..3
  const int wr   = wave >> 1;          // wave row in block (0..1)
  const int wc   = wave & 1;           // wave col in block (0..1)
  const int hl   = lane >> 4;
  const int l16  = lane & 15;

  const long rowBase = (long)blockIdx.y * 64 + wr * 32;   // 2 row frags
  const long colBase = (long)blockIdx.x * 128 + wc * 64;  // 4 col frags

  const unsigned short* __restrict__ xrow[2];
#pragma unroll
  for (int i = 0; i < 2; ++i) xrow[i] = X + (rowBase + i * 16 + l16) * K;
  const unsigned short* __restrict__ wrow[4];
#pragma unroll
  for (int j = 0; j < 4; ++j) wrow[j] = W + (colBase + j * 16 + l16) * K;

  Acc8 acc[2][4];
#pragma unroll
  for (int i = 0; i < 2; ++i)
#pragma unroll
    for (int j = 0; j < 4; ++j)
#pragma unroll
      for (int r = 0; r < 8; ++r) acc[i][j].f[r] = 0.0f;

  for (int k0 = 0; k0 < K; k0 += 32) {
    Frag16 a[2], b[4];
    // A 16x32 layout: e -> k = half*16 + e => one 32B chunk
#pragma unroll
    for (int i = 0; i < 2; ++i) {
      __builtin_prefetch((const void*)(xrow[i] + k0 + 512), 0, 3);
      a[i].q[0] = *(const v4u*)(xrow[i] + k0 + (hl << 4));
      a[i].q[1] = *(const v4u*)(xrow[i] + k0 + 8 + (hl << 4));
    }
    // B 32x16 layout: lane half selects K 16-block => one 32B chunk
#pragma unroll
    for (int j = 0; j < 4; ++j) {
      const v4u* bp = (const v4u*)(wrow[j] + k0 + (hl << 4));
      b[j].q[0] = bp[0];
      b[j].q[1] = bp[1];
    }
    // 8 independent WMMAs per k-step
#pragma unroll
    for (int i = 0; i < 2; ++i)
#pragma unroll
      for (int j = 0; j < 4; ++j)
        acc[i][j].v = __builtin_amdgcn_wmma_f32_16x16x32_bf16(
            false, a[i].v, false, b[j].v, (short)0, acc[i][j].v, false, false);
  }

  // C/D layout: lane = col n, VGPR r = row r + 8*half
#pragma unroll
  for (int i = 0; i < 2; ++i)
#pragma unroll
    for (int j = 0; j < 4; ++j)
#pragma unroll
      for (int r = 0; r < 8; ++r) {
        long row = rowBase + i * 16 + r + (hl << 3);
        long col = colBase + j * 16 + l16;
        float val = acc[i][j].f[r];
        if (outF)  outF[row * N + col] = val;
        if (outB)  outB[row * N + col] = f32_to_bf16_bits(val);
        if (outBT) outBT[col * M + row] = f32_to_bf16_bits(val);
      }
}

// ------------------- flash attention with diagonal mask --------------------
// Q/K: [B*L, H*64] bf16,  Vt: [H*64, B*L] bf16 (transposed by V projection).
// One wave per (b, h, 16-query tile); 32-key tiles; online softmax.
// K tile async-staged to LDS (double buffered).
#define HD 64
__global__ __launch_bounds__(32)
void flash_attn(const unsigned short* __restrict__ Q,
                const unsigned short* __restrict__ Km,
                const unsigned short* __restrict__ Vt,
                unsigned short* __restrict__ O,
                int L, int nheads, int Mtok)
{
  __shared__ __align__(16) unsigned short sK[2][32][64];  // 2 x 4KB key tiles
  __shared__ __align__(16) float          sS[16][32];     // masked scores
  __shared__ __align__(16) unsigned short sP[16][32];     // softmax num (bf16)
  __shared__ float sAlpha[16], sM[16], sL[16];

  const int lane = threadIdx.x & 31;
  const int hl   = lane >> 4;
  const int l16  = lane & 15;

  const int qt = blockIdx.x;
  const int h  = blockIdx.y;
  const int b  = blockIdx.z;
  const int D  = nheads * HD;                 // 1024
  const long base = (long)b * L * D;
  const int q0 = qt * 16;

  Acc8 o[4];
#pragma unroll
  for (int dg = 0; dg < 4; ++dg)
#pragma unroll
    for (int r = 0; r < 8; ++r) o[dg].f[r] = 0.0f;

  if (lane < 16) { sM[lane] = -1e30f; sL[lane] = 0.0f; }

  // Q A-fragments for head-dim k-steps 0 and 32
  Frag16 qa[2];
  const unsigned short* qrow = Q + base + (long)(q0 + l16) * D + h * HD;
#pragma unroll
  for (int s = 0; s < 2; ++s) {
    qa[s].q[0] = *(const v4u*)(qrow + s * 32 + (hl << 4));
    qa[s].q[1] = *(const v4u*)(qrow + s * 32 + 8 + (hl << 4));
  }

  // Vt columns for the 4 head-dim groups (lane = column within group)
  const unsigned short* __restrict__ vtcol[4];
#pragma unroll
  for (int dg = 0; dg < 4; ++dg)
    vtcol[dg] = Vt + (long)(h * HD + dg * 16 + l16) * Mtok + (long)b * L;

  // stage first K tile (each lane copies one 128B key row as 8x16B)
  {
    const unsigned short* ksrc = Km + base + (long)lane * D + h * HD;
#pragma unroll
    for (int j = 0; j < 8; ++j) async_copy16(ksrc + j * 8, &sK[0][lane][j * 8]);
  }

  for (int kt0 = 0; kt0 < L; kt0 += 32) {
    const int cur = (kt0 >> 5) & 1;
    wait_async_lds();
    __syncthreads();
    // kick off next tile's async copy while computing on current
    if (kt0 + 32 < L) {
      const unsigned short* ksrc =
          Km + base + (long)(kt0 + 32 + lane) * D + h * HD;
#pragma unroll
      for (int j = 0; j < 8; ++j)
        async_copy16(ksrc + j * 8, &sK[cur ^ 1][lane][j * 8]);
    }

    // ---- S = Q K^T : two 16x16 tiles over keys kt0..kt0+31 (from LDS) ----
    Acc8 sacc[2];
#pragma unroll
    for (int t = 0; t < 2; ++t) {
#pragma unroll
      for (int r = 0; r < 8; ++r) sacc[t].f[r] = 0.0f;
#pragma unroll
      for (int s = 0; s < 2; ++s) {
        Frag16 kb;   // lane = key n; lane-half selects K 16-block
        const v4u* kp = (const v4u*)&sK[cur][t * 16 + l16][s * 32 + (hl << 4)];
        kb.q[0] = kp[0];
        kb.q[1] = kp[1];
        sacc[t].v = __builtin_amdgcn_wmma_f32_16x16x32_bf16(
            false, qa[s].v, false, kb.v, (short)0, sacc[t].v, false, false);
      }
    }

    // ---- scale + diagonal mask, stage into LDS ----
#pragma unroll
    for (int t = 0; t < 2; ++t) {
      int colg = kt0 + t * 16 + l16;
#pragma unroll
      for (int r = 0; r < 8; ++r) {
        int rloc = r + (hl << 3);
        float sv = sacc[t].f[r] * 0.125f;          // 1/sqrt(64)
        if (colg == q0 + rloc) sv = -1e30f;        // mask self-attention
        sS[rloc][t * 16 + l16] = sv;
      }
    }
    __syncthreads();

    // ---- online softmax: lanes 0..15 each own one query row ----
    if (lane < 16) {
      float mold = sM[lane];
      float mnew = mold;
#pragma unroll
      for (int j = 0; j < 32; ++j) mnew = fmaxf(mnew, sS[lane][j]);
      float alpha = __expf(mold - mnew);
      float sum = 0.0f;
#pragma unroll
      for (int j = 0; j < 32; ++j) {
        float p = __expf(sS[lane][j] - mnew);
        sum += p;
        sP[lane][j] = f32_to_bf16_bits(p);
      }
      sM[lane]     = mnew;
      sL[lane]     = sL[lane] * alpha + sum;
      sAlpha[lane] = alpha;
    }
    __syncthreads();

    // ---- rescale O accumulators by alpha(row) ----
    float al[8];
#pragma unroll
    for (int r = 0; r < 8; ++r) al[r] = sAlpha[r + (hl << 3)];
#pragma unroll
    for (int dg = 0; dg < 4; ++dg)
#pragma unroll
      for (int r = 0; r < 8; ++r) o[dg].f[r] *= al[r];

    // ---- P as A-fragment (16 q-rows x 32 keys) from LDS ----
    Frag16 pa;
    pa.q[0] = *(const v4u*)&sP[l16][(hl << 4)];
    pa.q[1] = *(const v4u*)&sP[l16][8 + (hl << 4)];

    // ---- O += P * V : Vt gives contiguous key runs per head-dim column ----
#pragma unroll
    for (int dg = 0; dg < 4; ++dg) {
      Frag16 vb;   // lane = head-dim col; e -> key = kt0 + half*16 + e
      const v4u* vp = (const v4u*)(vtcol[dg] + kt0 + (hl << 4));
      vb.q[0] = vp[0];
      vb.q[1] = vp[1];
      o[dg].v = __builtin_amdgcn_wmma_f32_16x16x32_bf16(
          false, pa.v, false, vb.v, (short)0, o[dg].v, false, false);
    }
    __syncthreads();   // protect sS/sP reuse next iteration
  }

  // ---- normalize by running row sums and emit bf16 context ----
  float li[8];
#pragma unroll
  for (int r = 0; r < 8; ++r) li[r] = 1.0f / sL[r + (hl << 3)];
#pragma unroll
  for (int dg = 0; dg < 4; ++dg)
#pragma unroll
    for (int r = 0; r < 8; ++r) {
      int rowg = q0 + r + (hl << 3);
      float val = o[dg].f[r] * li[r];
      O[base + (long)rowg * D + h * HD + dg * 16 + l16] = f32_to_bf16_bits(val);
    }
}

// ---------------------------------------------------------------------------
extern "C" void kernel_launch(void* const* d_in, const int* in_sizes, int n_in,
                              void* d_out, int out_size, void* d_ws, size_t ws_size,
                              hipStream_t stream) {
  const float* x  = (const float*)d_in[0];
  const float* wq = (const float*)d_in[1];
  const float* wk = (const float*)d_in[2];
  const float* wv = (const float*)d_in[3];
  const float* wo = (const float*)d_in[4];

  const int Bsz = 4, L = 2048, D = 1024, H = 16;
  const int M = Bsz * L;                       // 8192 token rows

  size_t off = 0;
  auto carve = [&](size_t bytes) -> void* {
    void* p = (char*)d_ws + off;
    off += (bytes + 255) & ~(size_t)255;
    return p;
  };
  unsigned short* xb  = (unsigned short*)carve((size_t)M * D * 2);  // 16 MB
  unsigned short* wqb = (unsigned short*)carve((size_t)D * D * 2);  //  2 MB
  unsigned short* wkb = (unsigned short*)carve((size_t)D * D * 2);
  unsigned short* wvb = (unsigned short*)carve((size_t)D * D * 2);
  unsigned short* wob = (unsigned short*)carve((size_t)D * D * 2);
  unsigned short* Qb  = (unsigned short*)carve((size_t)M * D * 2);  // 16 MB
  unsigned short* Kb  = (unsigned short*)carve((size_t)M * D * 2);
  unsigned short* Vtb = (unsigned short*)carve((size_t)M * D * 2);  // [D][M]
  unsigned short* Cb  = (unsigned short*)carve((size_t)M * D * 2);  // context

  // 1) precision convert (vectorized x4)
  cvt_f32_bf16<<<(M * D / 4 + 255) / 256, 256, 0, stream>>>(x,  xb,  M * D / 4);
  cvt_f32_bf16<<<(D * D / 4 + 255) / 256, 256, 0, stream>>>(wq, wqb, D * D / 4);
  cvt_f32_bf16<<<(D * D / 4 + 255) / 256, 256, 0, stream>>>(wk, wkb, D * D / 4);
  cvt_f32_bf16<<<(D * D / 4 + 255) / 256, 256, 0, stream>>>(wv, wvb, D * D / 4);
  cvt_f32_bf16<<<(D * D / 4 + 255) / 256, 256, 0, stream>>>(wo, wob, D * D / 4);

  // 2) Q/K/V projections (V written transposed for contiguous PV fragments)
  dim3 gg(D / 128, M / 64);     // (8, 128) blocks of 128 threads
  wmma_gemm_xwT<<<gg, 128, 0, stream>>>(xb, wqb, nullptr, Qb, nullptr, M, D, D);
  wmma_gemm_xwT<<<gg, 128, 0, stream>>>(xb, wkb, nullptr, Kb, nullptr, M, D, D);
  wmma_gemm_xwT<<<gg, 128, 0, stream>>>(xb, wvb, nullptr, nullptr, Vtb, M, D, D);

  // 3) masked flash attention
  dim3 ga(L / 16, H, Bsz);      // (128, 16, 4) blocks of 1 wave
  flash_attn<<<ga, 32, 0, stream>>>(Qb, Kb, Vtb, Cb, L, H, M);

  // 4) output projection, f32 straight into d_out
  wmma_gemm_xwT<<<gg, 128, 0, stream>>>(Cb, wob, (float*)d_out, nullptr, nullptr,
                                        M, D, D);
}
